// RecurrentKDA_1090921693213
// MI455X (gfx1250) — matrix operand order repaired and checked by reading, not compiled, not run on hardware
//
#include <hip/hip_runtime.h>
#include <math.h>
#include <stdint.h>

typedef __bf16 bf16;
typedef __attribute__((ext_vector_type(8)))  float v8f;
typedef __attribute__((ext_vector_type(8)))  bf16  v8bf;
typedef __attribute__((ext_vector_type(16))) bf16  v16bf;

#define D_MODEL 1024
#define N_HEADS 16
#define BB      4
#define TT      4096
#define DK      64
#define MROWS   (BB * TT)          // 16384

// ---------------------------------------------------------------------------
// CDNA5 async global->LDS copy (ASYNCcnt path), 16B per lane.
// LDS address taken from the vdst VGPR; LDS offset = low 32 bits of generic pointer.
// ---------------------------------------------------------------------------
__device__ __forceinline__ void async_ld_b128(uint32_t lds_off, const void* g) {
  asm volatile("global_load_async_to_lds_b128 %0, %1, off"
               :: "v"(lds_off), "v"((unsigned long long)(uintptr_t)g)
               : "memory");
}
__device__ __forceinline__ void wait_async0() {
  asm volatile("s_wait_asynccnt 0x0" ::: "memory");
}

// ---------------------------------------------------------------------------
// LayerNorm: one block per row of 1024, output bf16
// ---------------------------------------------------------------------------
__global__ __launch_bounds__(256) void ln_kernel(const float* __restrict__ x,
                                                 const float* __restrict__ gamma,
                                                 const float* __restrict__ beta,
                                                 bf16* __restrict__ xn) {
  int row = blockIdx.x;
  int tid = threadIdx.x;
  const float* xr = x + (size_t)row * D_MODEL;
  float v[4];
#pragma unroll
  for (int i = 0; i < 4; ++i) v[i] = xr[tid + 256 * i];

  __shared__ float red[256];
  red[tid] = v[0] + v[1] + v[2] + v[3];
  __syncthreads();
  for (int off = 128; off > 0; off >>= 1) {
    if (tid < off) red[tid] += red[tid + off];
    __syncthreads();
  }
  float mean = red[0] * (1.0f / D_MODEL);
  __syncthreads();

  float s2 = 0.f;
#pragma unroll
  for (int i = 0; i < 4; ++i) {
    float d = v[i] - mean;
    s2 += d * d;
  }
  red[tid] = s2;
  __syncthreads();
  for (int off = 128; off > 0; off >>= 1) {
    if (tid < off) red[tid] += red[tid + off];
    __syncthreads();
  }
  float rstd = rsqrtf(red[0] * (1.0f / D_MODEL) + 1e-5f);

#pragma unroll
  for (int i = 0; i < 4; ++i) {
    int c = tid + 256 * i;
    float y = (v[i] - mean) * rstd * gamma[c] + beta[c];
    xn[(size_t)row * D_MODEL + c] = (bf16)y;
  }
}

// ---------------------------------------------------------------------------
// Transpose+convert: fp32 [K,N] -> bf16 [N,K]
// ---------------------------------------------------------------------------
__global__ __launch_bounds__(256) void transconv_kernel(const float* __restrict__ src,
                                                        bf16* __restrict__ dst,
                                                        int K, int N) {
  __shared__ float tile[32][33];
  int tid = threadIdx.x;
  int tx = tid & 31, ty = tid >> 5;      // 32 x 8
  int n0 = blockIdx.x * 32, k0 = blockIdx.y * 32;
#pragma unroll
  for (int yy = ty; yy < 32; yy += 8) {
    int kk = k0 + yy, nn = n0 + tx;
    tile[yy][tx] = (kk < K && nn < N) ? src[(size_t)kk * N + nn] : 0.f;
  }
  __syncthreads();
#pragma unroll
  for (int yy = ty; yy < 32; yy += 8) {
    int nn = n0 + yy, kk = k0 + tx;
    if (nn < N && kk < K) dst[(size_t)nn * K + kk] = (bf16)tile[tx][yy];
  }
}

// ---------------------------------------------------------------------------
// WMMA bf16 GEMM, M=16384, N=1024, K=1024.
//   A: bf16 [M,K] row-major.  Bt: bf16 [N,K] (K-contiguous).
//   mode 0: projections. grid.z picks weight/dest slot; z==3 gets sigmoid.
//           dest layout [T,B,H,dk] bf16.
//   mode 1: out = xres + A@Bt^T, fp32 row-major.
// Block: 256 thr (8 waves). Block tile 128(M) x 128(N). Wave tile 32 x 64.
// Double-buffered LDS, async global->LDS, one barrier per 32-K step.
// ---------------------------------------------------------------------------
__global__ __launch_bounds__(256) void gemm_kernel(const bf16* __restrict__ A,
                                                   const bf16* __restrict__ BtBase,
                                                   bf16* __restrict__ dstQKVG,
                                                   const float* __restrict__ xres,
                                                   float* __restrict__ outF,
                                                   int mode) {
  constexpr int K = D_MODEL;
  constexpr int LDP = 40;                 // padded LDS stride (80B: 16B aligned, conflict-free)
  __shared__ bf16 a_s[2][128 * LDP];
  __shared__ bf16 b_s[2][128 * LDP];

  int tid = threadIdx.x;
  int lane = tid & 31, wave = tid >> 5;
  int wm = wave & 3, wn = wave >> 2;      // 4 M-strips of 32, 2 N-strips of 64
  int bm = blockIdx.x * 128, bn = blockIdx.y * 128;

  const bf16* Bt;
  bf16* dst = nullptr;
  bool sig = false;
  if (mode == 0) {
    Bt = BtBase + (size_t)blockIdx.z * (D_MODEL * D_MODEL);
    dst = dstQKVG + (size_t)blockIdx.z * ((size_t)MROWS * D_MODEL);
    sig = (blockIdx.z == 3);
  } else {
    Bt = BtBase;
  }

  v8f acc[2][4] = {};

  // staging: 128 rows(cols) x 32 K, 2 lanes per row, 16 bf16 (32B) each
  int sr = tid >> 1, sh = tid & 1;
  const bf16* aG = A + (size_t)(bm + sr) * K + sh * 16;
  const bf16* bG = Bt + (size_t)(bn + sr) * K + sh * 16;
  uint32_t aL[2], bL[2];
#pragma unroll
  for (int p = 0; p < 2; ++p) {
    aL[p] = (uint32_t)(uintptr_t)(&a_s[p][sr * LDP + sh * 16]);
    bL[p] = (uint32_t)(uintptr_t)(&b_s[p][sr * LDP + sh * 16]);
  }

  int r16 = lane & 15;
  int akb = (lane < 16) ? 0 : 8;          // A frag K-base (ISA 16-bit A layout)
  int bkb = (lane < 16) ? 0 : 8;          // B frag K-base (same layout as A)

  // prologue: stage tile 0 into buffer 0
  async_ld_b128(aL[0],      aG);
  async_ld_b128(aL[0] + 16, aG + 8);
  async_ld_b128(bL[0],      bG);
  async_ld_b128(bL[0] + 16, bG + 8);
  wait_async0();
  __syncthreads();

  for (int k0 = 0; k0 < K; k0 += 32) {
    int cur = (k0 >> 5) & 1;
    if (k0 + 32 < K) {                    // stage next tile into other buffer
      int nxt = cur ^ 1;
      const bf16* aN = aG + k0 + 32;
      const bf16* bN = bG + k0 + 32;
      async_ld_b128(aL[nxt],      aN);
      async_ld_b128(aL[nxt] + 16, aN + 8);
      async_ld_b128(bL[nxt],      bN);
      async_ld_b128(bL[nxt] + 16, bN + 8);
    }

    // compute from current buffer
    const bf16* as = a_s[cur];
    const bf16* bs = b_s[cur];
#pragma unroll
    for (int mi = 0; mi < 2; ++mi) {
      const bf16* ap = as + (wm * 32 + mi * 16 + r16) * LDP;
      v8bf a_lo = *(const v8bf*)(ap + akb);
      v8bf a_hi = *(const v8bf*)(ap + akb + 16);
      v16bf af;
#pragma unroll
      for (int e = 0; e < 8; ++e) { af[e] = a_lo[e]; af[e + 8] = a_hi[e]; }
#pragma unroll
      for (int nt = 0; nt < 4; ++nt) {
        const bf16* bp = bs + (wn * 64 + nt * 16 + r16) * LDP + bkb;
        v8bf b_lo = *(const v8bf*)(bp);
        v8bf b_hi = *(const v8bf*)(bp + 16);
        v16bf bv;
#pragma unroll
        for (int e = 0; e < 8; ++e) { bv[e] = b_lo[e]; bv[e + 8] = b_hi[e]; }
        acc[mi][nt] = __builtin_amdgcn_wmma_f32_16x16x32_bf16(
            false, af, false, bv, (short)0, acc[mi][nt], false, false);
      }
    }

    wait_async0();        // next buffer fully written (this wave)
    __syncthreads();      // all waves: writes visible, reads of cur done
  }

  // Epilogue.  C/D layout: elem e, lanes 0-15 -> M=e, lanes 16-31 -> M=e+8; N=lane&15.
  int mofs = (lane >= 16) ? 8 : 0;
#pragma unroll
  for (int mi = 0; mi < 2; ++mi) {
#pragma unroll
    for (int nt = 0; nt < 4; ++nt) {
#pragma unroll
      for (int e = 0; e < 8; ++e) {
        int row = bm + wm * 32 + mi * 16 + mofs + e;
        int col = bn + wn * 64 + nt * 16 + r16;
        float val = acc[mi][nt][e];
        if (mode == 0) {
          if (sig) val = 1.0f / (1.0f + __expf(-val));
          int bb = row >> 12, tt = row & (TT - 1);
          int hh = col >> 6, dd = col & (DK - 1);
          dst[(((size_t)tt * BB + bb) * N_HEADS + hh) * DK + dd] = (bf16)val;
        } else {
          size_t idx = (size_t)row * D_MODEL + col;
          outF[idx] = xres[idx] + val;
        }
      }
    }
  }
}

// ---------------------------------------------------------------------------
// beta = sigmoid(xn @ Wbeta), one block per token row
// ---------------------------------------------------------------------------
__global__ __launch_bounds__(256) void beta_kernel(const bf16* __restrict__ xn,
                                                   const bf16* __restrict__ Wbt, // [16,1024]
                                                   float* __restrict__ betab) {  // [T,B,H]
  int m = blockIdx.x;
  int tid = threadIdx.x;
  __shared__ float xrow[D_MODEL];
#pragma unroll
  for (int i = 0; i < 4; ++i)
    xrow[tid + 256 * i] = (float)xn[(size_t)m * D_MODEL + tid + 256 * i];
  __syncthreads();

  int h = tid & 15, ks = tid >> 4;       // 16 heads x 16 k-slices of 64
  const bf16* w = Wbt + (size_t)h * D_MODEL + ks * 64;
  float acc = 0.f;
#pragma unroll 8
  for (int kk = 0; kk < 64; ++kk) acc += xrow[ks * 64 + kk] * (float)w[kk];

  __shared__ float pp[256];
  pp[tid] = acc;
  __syncthreads();
  if (tid < 16) {
    float s = 0.f;
#pragma unroll
    for (int k2 = 0; k2 < 16; ++k2) s += pp[k2 * 16 + tid];
    int bb = m >> 12, tt = m & (TT - 1);
    betab[((size_t)tt * BB + bb) * N_HEADS + tid] = 1.0f / (1.0f + __expf(-s));
  }
}

// ---------------------------------------------------------------------------
// Sequential delta-rule scan. One block per (b,h). State 64x64 fp32 in regs:
// thread (j = tid&63, rg = tid>>6) holds S[i][j] for i in [rg*16, rg*16+16).
//   S1 = S + beta*k v^T ; w_j = sum_i k_i a_i S1_ij
//   S2 = a o S1 - beta k w^T ; o_j = sum_i q_i S2_ij
// ---------------------------------------------------------------------------
__global__ __launch_bounds__(256) void scan_kernel(const bf16* __restrict__ qb,
                                                   const bf16* __restrict__ kb,
                                                   const bf16* __restrict__ vb,
                                                   const bf16* __restrict__ gb,
                                                   const float* __restrict__ betab,
                                                   bf16* __restrict__ obuf,
                                                   float* __restrict__ Sfinal) {
  int bh = blockIdx.x;
  int b = bh >> 4, h = bh & 15;
  int tid = threadIdx.x;
  int j = tid & 63, rg = tid >> 6;

  float S[16];
#pragma unroll
  for (int ii = 0; ii < 16; ++ii) S[ii] = 0.f;

  __shared__ float kS[64], vS[64], aS[64], qS[64];
  __shared__ float wpart[4][64], opart[4][64];
  __shared__ float betaS;

  for (int t = 0; t < TT; ++t) {
    size_t base = (((size_t)t * BB + b) * N_HEADS + h) * DK;
    if (tid < 64) {
      kS[tid] = (float)kb[base + tid];
      vS[tid] = (float)vb[base + tid];
      aS[tid] = (float)gb[base + tid];
      qS[tid] = (float)qb[base + tid];
    }
    if (tid == 0) betaS = betab[((size_t)t * BB + b) * N_HEADS + h];
    __syncthreads();

    float bt = betaS;
    float vj = vS[j];
    float wsum = 0.f;
#pragma unroll
    for (int ii = 0; ii < 16; ++ii) {
      int i = rg * 16 + ii;
      float ki = kS[i];
      float s1 = S[ii] + bt * ki * vj;
      float as1 = aS[i] * s1;
      S[ii] = as1;
      wsum += ki * as1;
    }
    wpart[rg][j] = wsum;
    __syncthreads();

    float w = wpart[0][j] + wpart[1][j] + wpart[2][j] + wpart[3][j];
    float osum = 0.f;
#pragma unroll
    for (int ii = 0; ii < 16; ++ii) {
      int i = rg * 16 + ii;
      float s2 = S[ii] - bt * kS[i] * w;
      S[ii] = s2;
      osum += qS[i] * s2;
    }
    opart[rg][j] = osum;
    __syncthreads();

    if (tid < 64) {
      float ov = opart[0][tid] + opart[1][tid] + opart[2][tid] + opart[3][tid];
      obuf[((size_t)b * TT + t) * D_MODEL + h * DK + tid] = (bf16)ov;
    }
    __syncthreads();
  }

#pragma unroll
  for (int ii = 0; ii < 16; ++ii) {
    int i = rg * 16 + ii;
    Sfinal[(((size_t)b * N_HEADS + h) * DK + i) * DK + j] = S[ii];
  }
}

// ---------------------------------------------------------------------------
// Launch
// ---------------------------------------------------------------------------
extern "C" void kernel_launch(void* const* d_in, const int* in_sizes, int n_in,
                              void* d_out, int out_size, void* d_ws, size_t ws_size,
                              hipStream_t stream) {
  (void)in_sizes; (void)n_in; (void)out_size; (void)ws_size;
  const float* x       = (const float*)d_in[0];
  const float* Wq      = (const float*)d_in[1];
  const float* Wk      = (const float*)d_in[2];
  const float* Wv      = (const float*)d_in[3];
  const float* Wg      = (const float*)d_in[4];
  const float* Wbeta   = (const float*)d_in[5];
  const float* Wo      = (const float*)d_in[6];
  const float* gamma   = (const float*)d_in[7];
  const float* beta_ln = (const float*)d_in[8];
  float* out = (float*)d_out;

  char* ws = (char*)d_ws;
  // workspace layout (bytes)
  bf16*  xn    = (bf16*)(ws + 0);             //  33,554,432
  bf16*  Wt    = (bf16*)(ws + 33554432);      //  10,485,760 (q,k,v,g,o transposed bf16)
  bf16*  Wbt   = (bf16*)(ws + 44040192);      //      32,768
  bf16*  qkvg  = (bf16*)(ws + 44072960);      // 134,217,728
  float* betab = (float*)(ws + 178290688);    //   1,048,576
  bf16*  obuf  = (bf16*)(ws + 179339264);     //  33,554,432
  // total: 212,893,696 B

  const size_t WSLOT = (size_t)D_MODEL * D_MODEL;     // 1,048,576 elems
  const size_t PSLOT = (size_t)MROWS * D_MODEL;       // 16,777,216 elems

  ln_kernel<<<MROWS, 256, 0, stream>>>(x, gamma, beta_ln, xn);

  transconv_kernel<<<dim3(32, 32), 256, 0, stream>>>(Wq, Wt + 0 * WSLOT, D_MODEL, D_MODEL);
  transconv_kernel<<<dim3(32, 32), 256, 0, stream>>>(Wk, Wt + 1 * WSLOT, D_MODEL, D_MODEL);
  transconv_kernel<<<dim3(32, 32), 256, 0, stream>>>(Wv, Wt + 2 * WSLOT, D_MODEL, D_MODEL);
  transconv_kernel<<<dim3(32, 32), 256, 0, stream>>>(Wg, Wt + 3 * WSLOT, D_MODEL, D_MODEL);
  transconv_kernel<<<dim3(32, 32), 256, 0, stream>>>(Wo, Wt + 4 * WSLOT, D_MODEL, D_MODEL);
  transconv_kernel<<<dim3(1, 32),  256, 0, stream>>>(Wbeta, Wbt, D_MODEL, N_HEADS);

  // q,k,v,g projections (fused, grid.z selects weight/dest, z==3 sigmoid)
  gemm_kernel<<<dim3(MROWS / 128, D_MODEL / 128, 4), 256, 0, stream>>>(
      xn, Wt, qkvg, nullptr, nullptr, 0);

  beta_kernel<<<MROWS, 256, 0, stream>>>(xn, Wbt, betab);

  float* Sfinal = out + PSLOT;
  scan_kernel<<<BB * N_HEADS, 256, 0, stream>>>(
      qkvg + 0 * PSLOT, qkvg + 1 * PSLOT, qkvg + 2 * PSLOT, qkvg + 3 * PSLOT,
      betab, obuf, Sfinal);

  // out = x + o @ Wo
  gemm_kernel<<<dim3(MROWS / 128, D_MODEL / 128, 1), 256, 0, stream>>>(
      obuf, Wt + 4 * WSLOT, nullptr, x, out, 1);
}
